// RETROBlock_55001351193159
// MI455X (gfx1250) — matrix operand order repaired and checked by reading, not compiled, not run on hardware
//
#include <hip/hip_runtime.h>
#include <hip/hip_bf16.h>
#include <math.h>

typedef unsigned short u16;
typedef __attribute__((ext_vector_type(8)))  float  v8f;
typedef __attribute__((ext_vector_type(16))) __bf16 v16bf;

union Frag16 {            // 32 bytes = one v16bf WMMA operand (8 VGPRs)
    uint4 q[2];
    v16bf v;
};

__device__ __forceinline__ u16 f2bf(float f) {
    unsigned int u = __float_as_uint(f);
    unsigned int r = u + 0x7FFFu + ((u >> 16) & 1u);   // round-to-nearest-even
    return (u16)(r >> 16);
}

// ---------------------------------------------------------------------------
// CDNA5 async global->LDS staging (ASYNCcnt-tracked). The builtin takes
// (int4 addrspace(1)* src, int4 addrspace(3)* dst, imm offset, imm cpol).
// Falls back to regular b128 load + ds_store staging if not declared.
// ---------------------------------------------------------------------------
#if defined(__has_builtin)
#  if __has_builtin(__builtin_amdgcn_global_load_async_to_lds_b128)
#    define HAVE_ASYNC_LDS 1
#  endif
#endif
#ifndef HAVE_ASYNC_LDS
#  define HAVE_ASYNC_LDS 0
#endif

#if HAVE_ASYNC_LDS
#define GLOBAL_AS __attribute__((address_space(1)))
#define LDS_AS    __attribute__((address_space(3)))
typedef int v4i_b128 __attribute__((vector_size(4 * sizeof(int))));
__device__ __forceinline__ void async_copy_b128(const u16* gsrc, u16* ldst) {
    __builtin_amdgcn_global_load_async_to_lds_b128(
        (GLOBAL_AS v4i_b128*)(GLOBAL_AS void*)gsrc,
        (LDS_AS    v4i_b128*)(LDS_AS    void*)ldst, 0, 0);
}
__device__ __forceinline__ void async_wait0() {
#  if __has_builtin(__builtin_amdgcn_s_wait_asynccnt)
    __builtin_amdgcn_s_wait_asynccnt(0);
#  else
    asm volatile("s_wait_asynccnt 0" ::: "memory");
#  endif
}
#endif

// ---------------------------------------------------------------------------
// Weight convert: f32 [R][C] -> bf16 transposed [C][R]
// ---------------------------------------------------------------------------
__global__ __launch_bounds__(256) void transpose_f32_bf16(
    const float* __restrict__ in, u16* __restrict__ out, int R, int C)
{
    __shared__ float tile[32][33];
    int bx = blockIdx.x * 32;          // col base in 'in'
    int by = blockIdx.y * 32;          // row base in 'in'
    int tx = threadIdx.x, ty = threadIdx.y;   // 32 x 8
#pragma unroll
    for (int i = 0; i < 32; i += 8)
        tile[ty + i][tx] = in[(size_t)(by + ty + i) * C + bx + tx];
    __syncthreads();
#pragma unroll
    for (int i = 0; i < 32; i += 8)
        out[(size_t)(bx + ty + i) * R + by + tx] = f2bf(tile[tx][ty + i]);
}

// ---------------------------------------------------------------------------
// LayerNorm over D=1024, f32 in -> bf16 out. One 256-thread block per row.
// ---------------------------------------------------------------------------
__global__ __launch_bounds__(256) void layernorm_bf16(
    const float* __restrict__ x, const float* __restrict__ g,
    const float* __restrict__ b, u16* __restrict__ out)
{
    const int D = 1024;
    int row = blockIdx.x;
    int tid = threadIdx.x;
    const float* xr = x + (size_t)row * D;
    float4 v = *(const float4*)&xr[tid * 4];
    float s  = v.x + v.y + v.z + v.w;
    float sq = v.x * v.x + v.y * v.y + v.z * v.z + v.w * v.w;
#pragma unroll
    for (int m = 1; m < 32; m <<= 1) {
        s  += __shfl_xor(s,  m, 32);
        sq += __shfl_xor(sq, m, 32);
    }
    __shared__ float red[16];
    int wave = tid >> 5, lane = tid & 31;
    if (lane == 0) { red[wave] = s; red[8 + wave] = sq; }
    __syncthreads();
    float ts = 0.f, tq = 0.f;
#pragma unroll
    for (int i = 0; i < 8; ++i) { ts += red[i]; tq += red[8 + i]; }
    float mean = ts * (1.f / D);
    float var  = tq * (1.f / D) - mean * mean;
    float rstd = rsqrtf(var + 1e-5f);
    float4 gv = *(const float4*)&g[tid * 4];
    float4 bv = *(const float4*)&b[tid * 4];
    unsigned int p0 = (unsigned int)f2bf((v.x - mean) * rstd * gv.x + bv.x)
                    | ((unsigned int)f2bf((v.y - mean) * rstd * gv.y + bv.y) << 16);
    unsigned int p1 = (unsigned int)f2bf((v.z - mean) * rstd * gv.z + bv.z)
                    | ((unsigned int)f2bf((v.w - mean) * rstd * gv.w + bv.w) << 16);
    uint2 pk; pk.x = p0; pk.y = p1;
    *(uint2*)&out[(size_t)row * D + tid * 4] = pk;
}

// ---------------------------------------------------------------------------
// BF16 WMMA GEMM: C[M,N] = A[M,K](bf16) * Bt[N,K](bf16, pre-transposed)
// Tiles: 128x128x32. 256 threads = 8 waves (4 along M x 2 along N);
// each wave: 32x64 patch = 2x4 tiles of 16x16 -> 8 x v_wmma per k-step.
// Epilogue: +residual (f32), optional exact GELU, f32 and/or bf16 stores.
// ---------------------------------------------------------------------------
__global__ __launch_bounds__(256) void gemm_bf16_wmma(
    const u16* __restrict__ A, const u16* __restrict__ Bt,
    const float* __restrict__ residual, float* __restrict__ outF,
    u16* __restrict__ outB, int M, int N, int K, int gelu)
{
    __shared__ __attribute__((aligned(16))) u16 As[128 * 40];   // [m][k], pad 32->40
    __shared__ __attribute__((aligned(16))) u16 Bs[128 * 40];   // [n][k], pad 32->40

    const int tid  = threadIdx.x;
    const int lane = tid & 31;
    const int wave = tid >> 5;
    const int wm   = wave & 3;          // 0..3 -> 32-row slab
    const int wn   = wave >> 2;         // 0..1 -> 64-col slab
    const int half = lane >> 4;         // K-half select per WMMA layout
    const int l16  = lane & 15;

    const int m0 = blockIdx.y * 128;
    const int n0 = blockIdx.x * 128;

    const int ldr_row  = tid & 127;
    const int ldr_half = tid >> 7;

    v8f acc[2][4];
#pragma unroll
    for (int mt = 0; mt < 2; ++mt)
#pragma unroll
        for (int nt = 0; nt < 4; ++nt)
            acc[mt][nt] = (v8f){0.f,0.f,0.f,0.f,0.f,0.f,0.f,0.f};

    for (int kt = 0; kt < K; kt += 32) {
        __syncthreads();
        {
            const u16* gA = A  + (size_t)(m0 + ldr_row) * K + kt + ldr_half * 16;
            const u16* gB = Bt + (size_t)(n0 + ldr_row) * K + kt + ldr_half * 16;
            u16* lA = &As[ldr_row * 40 + ldr_half * 16];
            u16* lB = &Bs[ldr_row * 40 + ldr_half * 16];
            if (kt + 32 < K) {  // global_prefetch_b8, WGP scope (locality 3)
                __builtin_prefetch(gA + 32, 0, 3);
                __builtin_prefetch(gB + 32, 0, 3);
            }
#if HAVE_ASYNC_LDS
            async_copy_b128(gA,     lA);
            async_copy_b128(gA + 8, lA + 8);
            async_copy_b128(gB,     lB);
            async_copy_b128(gB + 8, lB + 8);
            async_wait0();
#else
            uint4 a0 = ((const uint4*)gA)[0], a1 = ((const uint4*)gA)[1];
            uint4 b0 = ((const uint4*)gB)[0], b1 = ((const uint4*)gB)[1];
            ((uint4*)lA)[0] = a0; ((uint4*)lA)[1] = a1;
            ((uint4*)lB)[0] = b0; ((uint4*)lB)[1] = b1;
#endif
        }
        __syncthreads();

        Frag16 afr[2], bfr[4];
#pragma unroll
        for (int mt = 0; mt < 2; ++mt) {        // A 16x32: lanes<16 K0-7 & 16-23
            int row = wm * 32 + mt * 16 + l16;
            afr[mt].q[0] = *(const uint4*)&As[row * 40 + half * 8];
            afr[mt].q[1] = *(const uint4*)&As[row * 40 + 16 + half * 8];
        }
#pragma unroll
        for (int nt = 0; nt < 4; ++nt) {        // B 32x16: lanes<16 K0-7 & 16-23
            int col = wn * 64 + nt * 16 + l16;
            bfr[nt].q[0] = *(const uint4*)&Bs[col * 40 + half * 8];
            bfr[nt].q[1] = *(const uint4*)&Bs[col * 40 + 16 + half * 8];
        }
#pragma unroll
        for (int mt = 0; mt < 2; ++mt)
#pragma unroll
            for (int nt = 0; nt < 4; ++nt)
                acc[mt][nt] = __builtin_amdgcn_wmma_f32_16x16x32_bf16(
                    false, afr[mt].v, false, bfr[nt].v,
                    (short)0, acc[mt][nt], false, false);
    }

#pragma unroll
    for (int mt = 0; mt < 2; ++mt)
#pragma unroll
        for (int nt = 0; nt < 4; ++nt) {
            int gc = n0 + wn * 64 + nt * 16 + l16;
#pragma unroll
            for (int r = 0; r < 8; ++r) {
                int row = m0 + wm * 32 + mt * 16 + r + half * 8;
                float v = acc[mt][nt][r];
                if (residual) v += residual[(size_t)row * N + gc];
                if (gelu) v = 0.5f * v * (1.f + erff(v * 0.70710678f));
                if (outF) outF[(size_t)row * N + gc] = v;
                if (outB) outB[(size_t)row * N + gc] = f2bf(v);
            }
        }
}

// ---------------------------------------------------------------------------
// Flash attention (WMMA, bf16 QKV, f32 online softmax). H=16, d=64, ld=1024.
// Block = 128 threads = 4 waves; each wave owns 16 query rows.
// Key/value streamed in blocks of 32 via LDS (V stored transposed).
// ---------------------------------------------------------------------------
__global__ __launch_bounds__(128) void flash_wmma(
    const u16* __restrict__ Qg, const u16* __restrict__ Kg,
    const u16* __restrict__ Vg, u16* __restrict__ Og,
    int q_batch_rows, int kv_batch_rows, int Tk, int causal, float scale)
{
    const int D = 1024;
    __shared__ __attribute__((aligned(16))) u16 Ks [32 * 72];      // [key][dim], pad 64->72
    __shared__ __attribute__((aligned(16))) u16 Vst[64 * 40];      // [dim][key], pad 32->40
    __shared__ __attribute__((aligned(16))) u16 Ps [4 * 16 * 40];  // per-wave P tile

    const int tid  = threadIdx.x;
    const int lane = tid & 31;
    const int wave = tid >> 5;           // 0..3
    const int half = lane >> 4;
    const int l16  = lane & 15;

    const int h      = blockIdx.y;
    const int batch  = blockIdx.z;
    const int qb     = blockIdx.x;
    const int qrow0  = batch * q_batch_rows + qb * 64;
    const int kvrow0 = batch * kv_batch_rows;
    const int qloc0  = qb * 64;
    const int hoff   = h * 64;

    // Preload Q fragments (16 rows x 64 dims -> two 16x32 A-fragments)
    Frag16 qf[2];
    {
        const u16* qp = Qg + (size_t)(qrow0 + wave * 16 + l16) * D + hoff;
#pragma unroll
        for (int c = 0; c < 2; ++c) {
            qf[c].q[0] = *(const uint4*)&qp[c * 32 + half * 8];
            qf[c].q[1] = *(const uint4*)&qp[c * 32 + half * 8 + 16];
        }
    }

    float m_i[8], l_i[8];
    v8f o[4];
#pragma unroll
    for (int r = 0; r < 8; ++r) { m_i[r] = -INFINITY; l_i[r] = 0.f; }
#pragma unroll
    for (int nd = 0; nd < 4; ++nd) o[nd] = (v8f){0.f,0.f,0.f,0.f,0.f,0.f,0.f,0.f};

    const int kend = causal ? (qloc0 + 64) : Tk;

    for (int kb = 0; kb < kend; kb += 32) {
        __syncthreads();
        {   // cooperative K/V stage: thread -> (key, 16-dim slice)
            int key  = tid >> 2;
            int doff = (tid & 3) * 16;
            const u16* kp = Kg + (size_t)(kvrow0 + kb + key) * D + hoff + doff;
            const u16* vp = Vg + (size_t)(kvrow0 + kb + key) * D + hoff + doff;
#if HAVE_ASYNC_LDS
            async_copy_b128(kp,     &Ks[key * 72 + doff]);
            async_copy_b128(kp + 8, &Ks[key * 72 + doff + 8]);
#else
            uint4 k0 = *(const uint4*)kp, k1 = *(const uint4*)(kp + 8);
            uint4* kd = (uint4*)&Ks[key * 72 + doff];
            kd[0] = k0; kd[1] = k1;
#endif
            uint4 v0 = *(const uint4*)vp, v1 = *(const uint4*)(vp + 8);
            const u16* va = (const u16*)&v0;
            const u16* vb = (const u16*)&v1;
#pragma unroll
            for (int i = 0; i < 8; ++i) Vst[(doff + i)     * 40 + key] = va[i];
#pragma unroll
            for (int i = 0; i < 8; ++i) Vst[(doff + 8 + i) * 40 + key] = vb[i];
#if HAVE_ASYNC_LDS
            async_wait0();
#endif
        }
        __syncthreads();

        // S = Q * K^T  (two 16x16 score tiles over 32 keys)
        v8f s[2];
        s[0] = (v8f){0.f,0.f,0.f,0.f,0.f,0.f,0.f,0.f};
        s[1] = (v8f){0.f,0.f,0.f,0.f,0.f,0.f,0.f,0.f};
#pragma unroll
        for (int nt = 0; nt < 2; ++nt) {
#pragma unroll
            for (int c = 0; c < 2; ++c) {
                Frag16 bf;
                int key     = nt * 16 + l16;
                int dimbase = c * 32 + half * 8;
                bf.q[0] = *(const uint4*)&Ks[key * 72 + dimbase];
                bf.q[1] = *(const uint4*)&Ks[key * 72 + dimbase + 16];
                s[nt] = __builtin_amdgcn_wmma_f32_16x16x32_bf16(
                    false, qf[c].v, false, bf.v, (short)0, s[nt], false, false);
            }
        }

        // scale + causal mask
        const int qrl = qloc0 + wave * 16 + half * 8;
#pragma unroll
        for (int nt = 0; nt < 2; ++nt)
#pragma unroll
            for (int r = 0; r < 8; ++r) {
                float sv = s[nt][r] * scale;
                if (causal && (kb + nt * 16 + l16) > (qrl + r)) sv = -INFINITY;
                s[nt][r] = sv;
            }

        // online softmax (row reductions across 16-lane groups)
        float alpha_r[8];
#pragma unroll
        for (int r = 0; r < 8; ++r) {
            float mx = fmaxf(s[0][r], s[1][r]);
#pragma unroll
            for (int m = 1; m < 16; m <<= 1) mx = fmaxf(mx, __shfl_xor(mx, m, 32));
            float mnew  = fmaxf(m_i[r], mx);
            float msafe = (mnew == -INFINITY) ? 0.f : mnew;
            float a  = __expf(m_i[r]  - msafe);
            float p0 = __expf(s[0][r] - msafe);
            float p1 = __expf(s[1][r] - msafe);
            float rs = p0 + p1;
#pragma unroll
            for (int m = 1; m < 16; m <<= 1) rs += __shfl_xor(rs, m, 32);
            l_i[r] = l_i[r] * a + rs;
            m_i[r] = mnew;
            alpha_r[r] = a;
            int prow = r + half * 8;
            Ps[wave * 640 + prow * 40 + l16]      = f2bf(p0);
            Ps[wave * 640 + prow * 40 + 16 + l16] = f2bf(p1);
        }
#pragma unroll
        for (int nd = 0; nd < 4; ++nd)
#pragma unroll
            for (int r = 0; r < 8; ++r) o[nd][r] *= alpha_r[r];

        __syncthreads();

        // O += P * V
        Frag16 pf;
        pf.q[0] = *(const uint4*)&Ps[wave * 640 + l16 * 40 + half * 8];
        pf.q[1] = *(const uint4*)&Ps[wave * 640 + l16 * 40 + 16 + half * 8];
#pragma unroll
        for (int nd = 0; nd < 4; ++nd) {
            Frag16 vf;
            int dim = nd * 16 + l16;
            vf.q[0] = *(const uint4*)&Vst[dim * 40 + half * 8];
            vf.q[1] = *(const uint4*)&Vst[dim * 40 + 16 + half * 8];
            o[nd] = __builtin_amdgcn_wmma_f32_16x16x32_bf16(
                false, pf.v, false, vf.v, (short)0, o[nd], false, false);
        }
    }

    // normalize + store bf16
#pragma unroll
    for (int nd = 0; nd < 4; ++nd)
#pragma unroll
        for (int r = 0; r < 8; ++r) {
            int row = qrow0 + wave * 16 + r + half * 8;
            float ov = o[nd][r] / l_i[r];
            Og[(size_t)row * D + hoff + nd * 16 + l16] = f2bf(ov);
        }
}

// ---------------------------------------------------------------------------
// Host-side orchestration
// ---------------------------------------------------------------------------
extern "C" void kernel_launch(void* const* d_in, const int* in_sizes, int n_in,
                              void* d_out, int out_size, void* d_ws, size_t ws_size,
                              hipStream_t stream) {
    (void)in_sizes; (void)n_in; (void)out_size; (void)ws_size;

    const float* x        = (const float*)d_in[0];
    const float* neigh    = (const float*)d_in[1];
    const float* sa_ln_g  = (const float*)d_in[2];
    const float* sa_ln_b  = (const float*)d_in[3];
    const float* sa_wq    = (const float*)d_in[4];
    const float* sa_wk    = (const float*)d_in[5];
    const float* sa_wv    = (const float*)d_in[6];
    const float* sa_wo    = (const float*)d_in[7];
    const float* clnq_g   = (const float*)d_in[8];
    const float* clnq_b   = (const float*)d_in[9];
    const float* clnkv_g  = (const float*)d_in[10];
    const float* clnkv_b  = (const float*)d_in[11];
    const float* cca_wq   = (const float*)d_in[12];
    const float* cca_wk   = (const float*)d_in[13];
    const float* cca_wv   = (const float*)d_in[14];
    const float* cca_wo   = (const float*)d_in[15];
    const float* fln_g    = (const float*)d_in[16];
    const float* fln_b    = (const float*)d_in[17];
    const float* ffn_w1   = (const float*)d_in[18];
    const float* ffn_w2   = (const float*)d_in[19];
    // d_in[20] = chunk_size (64, constant here)

    const int M1 = 4096;        // B*T rows
    const int D  = 1024;
    const int MK = 16384;       // neighbor rows (B*NC*K*NL)
    const int DF = 4096;

    char* ws = (char*)d_ws;
    size_t off = 0;
    auto alloc = [&](size_t bytes) -> void* {
        void* p = (void*)(ws + off);
        off += (bytes + 255) & ~(size_t)255;
        return p;
    };

    u16* wq_t  = (u16*)alloc((size_t)D * D * 2);
    u16* wk_t  = (u16*)alloc((size_t)D * D * 2);
    u16* wv_t  = (u16*)alloc((size_t)D * D * 2);
    u16* wo_t  = (u16*)alloc((size_t)D * D * 2);
    u16* cwq_t = (u16*)alloc((size_t)D * D * 2);
    u16* cwk_t = (u16*)alloc((size_t)D * D * 2);
    u16* cwv_t = (u16*)alloc((size_t)D * D * 2);
    u16* cwo_t = (u16*)alloc((size_t)D * D * 2);
    u16* w1_t  = (u16*)alloc((size_t)D * DF * 2);
    u16* w2_t  = (u16*)alloc((size_t)DF * D * 2);
    u16* bufA    = (u16*)alloc((size_t)M1 * D * 2);   // h / qn / h2
    u16* bufQ    = (u16*)alloc((size_t)M1 * D * 2);   // q / cq
    u16* bufK    = (u16*)alloc((size_t)MK * D * 2);   // k / ck
    u16* bufV    = (u16*)alloc((size_t)MK * D * 2);   // v / cv
    u16* bufAttn = (u16*)alloc((size_t)M1 * D * 2);   // attn / cca-attn
    float* x1    = (float*)alloc((size_t)M1 * D * 4);
    float* x2    = (float*)alloc((size_t)M1 * D * 4);
    u16* bufKVN  = (u16*)alloc((size_t)MK * D * 2);   // kvn, then FFN mid (M1 x DF)

    dim3 tb(32, 8);
    transpose_f32_bf16<<<dim3(D/32, D/32),  tb, 0, stream>>>(sa_wq,  wq_t,  D, D);
    transpose_f32_bf16<<<dim3(D/32, D/32),  tb, 0, stream>>>(sa_wk,  wk_t,  D, D);
    transpose_f32_bf16<<<dim3(D/32, D/32),  tb, 0, stream>>>(sa_wv,  wv_t,  D, D);
    transpose_f32_bf16<<<dim3(D/32, D/32),  tb, 0, stream>>>(sa_wo,  wo_t,  D, D);
    transpose_f32_bf16<<<dim3(D/32, D/32),  tb, 0, stream>>>(cca_wq, cwq_t, D, D);
    transpose_f32_bf16<<<dim3(D/32, D/32),  tb, 0, stream>>>(cca_wk, cwk_t, D, D);
    transpose_f32_bf16<<<dim3(D/32, D/32),  tb, 0, stream>>>(cca_wv, cwv_t, D, D);
    transpose_f32_bf16<<<dim3(D/32, D/32),  tb, 0, stream>>>(cca_wo, cwo_t, D, D);
    transpose_f32_bf16<<<dim3(DF/32, D/32), tb, 0, stream>>>(ffn_w1, w1_t,  D, DF);
    transpose_f32_bf16<<<dim3(D/32, DF/32), tb, 0, stream>>>(ffn_w2, w2_t,  DF, D);

    // ---- self attention ----
    layernorm_bf16<<<M1, 256, 0, stream>>>(x, sa_ln_g, sa_ln_b, bufA);
    gemm_bf16_wmma<<<dim3(D/128, M1/128), 256, 0, stream>>>(bufA, wq_t, nullptr, nullptr, bufQ, M1, D, D, 0);
    gemm_bf16_wmma<<<dim3(D/128, M1/128), 256, 0, stream>>>(bufA, wk_t, nullptr, nullptr, bufK, M1, D, D, 0);
    gemm_bf16_wmma<<<dim3(D/128, M1/128), 256, 0, stream>>>(bufA, wv_t, nullptr, nullptr, bufV, M1, D, D, 0);
    flash_wmma<<<dim3(32, 16, 2), 128, 0, stream>>>(bufQ, bufK, bufV, bufAttn,
                                                    2048, 2048, 2048, 1, 0.125f);
    gemm_bf16_wmma<<<dim3(D/128, M1/128), 256, 0, stream>>>(bufAttn, wo_t, x, x1, nullptr, M1, D, D, 0);

    // ---- chunked cross attention (padded_T == T, pure reshape) ----
    layernorm_bf16<<<M1, 256, 0, stream>>>(x1, clnq_g, clnq_b, bufA);
    layernorm_bf16<<<MK, 256, 0, stream>>>(neigh, clnkv_g, clnkv_b, bufKVN);
    gemm_bf16_wmma<<<dim3(D/128, M1/128), 256, 0, stream>>>(bufA,   cwq_t, nullptr, nullptr, bufQ, M1, D, D, 0);
    gemm_bf16_wmma<<<dim3(D/128, MK/128), 256, 0, stream>>>(bufKVN, cwk_t, nullptr, nullptr, bufK, MK, D, D, 0);
    gemm_bf16_wmma<<<dim3(D/128, MK/128), 256, 0, stream>>>(bufKVN, cwv_t, nullptr, nullptr, bufV, MK, D, D, 0);
    flash_wmma<<<dim3(1, 16, 64), 128, 0, stream>>>(bufQ, bufK, bufV, bufAttn,
                                                    64, 256, 256, 0, 0.125f);
    gemm_bf16_wmma<<<dim3(D/128, M1/128), 256, 0, stream>>>(bufAttn, cwo_t, x1, x2, nullptr, M1, D, D, 0);

    // ---- FFN ----
    layernorm_bf16<<<M1, 256, 0, stream>>>(x2, fln_g, fln_b, bufA);
    gemm_bf16_wmma<<<dim3(DF/128, M1/128), 256, 0, stream>>>(bufA, w1_t, nullptr, nullptr, bufKVN, M1, DF, D, 1);
    gemm_bf16_wmma<<<dim3(D/128, M1/128), 256, 0, stream>>>(bufKVN, w2_t, x2, (float*)d_out, nullptr, M1, D, DF, 0);
}
